// RecurrentRGCN_21741124452464
// MI455X (gfx1250) — matrix operand order repaired and checked by reading, not compiled, not run on hardware
//
#include <hip/hip_runtime.h>
#include <hip/hip_bf16.h>
#include <math.h>

// ---------------------------------------------------------------------------
// RecurrentRGCN for MI455X (gfx1250, wave32, WMMA bf16 16x16x32)
// N=50000, H=128, R2=400, T=3, E=500000
// ---------------------------------------------------------------------------

typedef __attribute__((ext_vector_type(16))) __bf16 v16bf;
typedef __attribute__((ext_vector_type(8)))  __bf16 v8bf;
typedef __attribute__((ext_vector_type(8)))  float  v8f;

#define HDIM 128
#define SLOPE_F 0.22916666666666666f   // (1/8 + 1/3)/2

__device__ __forceinline__ __bf16 tobf(float x) { return (__bf16)x; }

// ---------------------------------------------------------------------------
// Weight prep: pack f32 weights into bf16 [O][K] row-major (optionally
// transposing a [K][O] source). Tiny (<= 384x256), runs once per launch.
// ---------------------------------------------------------------------------
__global__ void k_prep_w(const float* __restrict__ in, __bf16* __restrict__ out,
                         int O, int K, int ld, int transpose)
{
  int i = blockIdx.x * blockDim.x + threadIdx.x;
  if (i >= O * K) return;
  int o = i / K, k = i - o * K;
  float v = transpose ? in[(size_t)k * ld + o] : in[(size_t)o * ld + k];
  out[i] = tobf(v);
}

// ---------------------------------------------------------------------------
// GEMM: C[m,o] = sum_k A(m,k) * Bt[o][k] (+ bias[o])
//   A (f32) split at K0 (multiple of 32): k<K0 -> A0, else A1 (col k-K0).
//   Bt: packed bf16 [O][K] row-major (pre-transposed weights).
// Requirements: M%16==0, O%64==0, K%32==0, K0%32==0.
// One wave -> 16 rows x 64 cols (4 accumulators), 4 waves / block.
// ---------------------------------------------------------------------------
__global__ __launch_bounds__(128) void k_gemm_bf16(
    const float* __restrict__ A0, const float* __restrict__ A1, int K0,
    int lda0, int lda1,
    const __bf16* __restrict__ Bt,
    const float* __restrict__ bias,
    float* __restrict__ C, int ldc,
    int M, int O, int K)
{
  const int wave = threadIdx.x >> 5;
  const int lane = threadIdx.x & 31;
  const int mTile = (blockIdx.x * 4 + wave) * 16;
  const int oTile = blockIdx.y * 64;
  if (mTile >= M) return;                 // wave-uniform: EXEC all-ones below

  const int half = lane >> 4;             // half-wave select
  const int lm   = lane & 15;

  // per-lane base pointers (computed once)
  const int arow = mTile + lm;
  const float* __restrict__ Arow0 = A0 + (size_t)arow * lda0;
  const float* __restrict__ Arow1 = A1 ? (A1 + (size_t)arow * lda1 - K0) : Arow0;
  const __bf16* __restrict__ bp0 = Bt + (size_t)(oTile      + lm) * K + half * 16;
  const __bf16* __restrict__ bp1 = Bt + (size_t)(oTile + 16 + lm) * K + half * 16;
  const __bf16* __restrict__ bp2 = Bt + (size_t)(oTile + 32 + lm) * K + half * 16;
  const __bf16* __restrict__ bp3 = Bt + (size_t)(oTile + 48 + lm) * K + half * 16;

  v8f acc[4] = {};

  for (int k0 = 0; k0 < K; k0 += 32) {
    // ---- A fragment (16x32 bf16): K = k0+half*16+{0..15}
    //      whole 32-block lies in A0 or A1 (K0 % 32 == 0) -> scalar select
    const float* ap = ((k0 < K0) ? Arow0 : Arow1) + k0 + half * 16;
    float tmp[16];
    *(float4*)(tmp + 0)  = *(const float4*)(ap + 0);
    *(float4*)(tmp + 4)  = *(const float4*)(ap + 4);
    *(float4*)(tmp + 8)  = *(const float4*)(ap + 8);
    *(float4*)(tmp + 12) = *(const float4*)(ap + 12);
    v16bf a;
    #pragma unroll
    for (int j = 0; j < 16; ++j) a[j] = tobf(tmp[j]);

    // ---- 4 B fragments (32x16 bf16): contiguous bf16, two b128 loads each
    #pragma unroll
    for (int t = 0; t < 4; ++t) {
      const __bf16* bp = (t == 0 ? bp0 : t == 1 ? bp1 : t == 2 ? bp2 : bp3) + k0;
      v8bf blo = *(const v8bf*)(bp);
      v8bf bhi = *(const v8bf*)(bp + 8);
      v16bf b = __builtin_shufflevector(blo, bhi, 0, 1, 2, 3, 4, 5, 6, 7,
                                        8, 9, 10, 11, 12, 13, 14, 15);
      acc[t] = __builtin_amdgcn_wmma_f32_16x16x32_bf16(
          false, a, false, b, (short)0, acc[t], false, false);
    }
  }

  // ---- store: VGPR r -> row mTile + r + 8*half, col = oTile + t*16 + lm
  #pragma unroll
  for (int t = 0; t < 4; ++t) {
    const int col = oTile + t * 16 + lm;
    const float bv = bias ? bias[col] : 0.0f;
    #pragma unroll
    for (int r = 0; r < 8; ++r) {
      const int rowm = mTile + r + 8 * half;
      C[(size_t)rowm * ldc + col] = acc[t][r] + bv;
    }
  }
}

// ---------------------------------------------------------------------------
// Elementwise / reduction kernels (wave32-aware)
// ---------------------------------------------------------------------------
__global__ void k_fill(float* __restrict__ p, int n, float v) {
  int i = blockIdx.x * blockDim.x + threadIdx.x;
  if (i < n) p[i] = v;
}

// out[row] = in[row] / max(||in[row]||, 1e-12), row length 128, wave per row
__global__ __launch_bounds__(256) void k_l2norm(
    const float* __restrict__ in, float* __restrict__ out, int rows)
{
  const int wave = threadIdx.x >> 5, lane = threadIdx.x & 31;
  const int row = blockIdx.x * 8 + wave;
  if (row >= rows) return;
  float4 v = ((const float4*)(in + (size_t)row * HDIM))[lane];
  float ss = v.x * v.x + v.y * v.y + v.z * v.z + v.w * v.w;
  #pragma unroll
  for (int off = 16; off; off >>= 1) ss += __shfl_xor(ss, off, 32);
  const float s = 1.f / fmaxf(sqrtf(ss), 1e-12f);
  float4 o; o.x = v.x * s; o.y = v.y * s; o.z = v.z * s; o.w = v.w * s;
  ((float4*)(out + (size_t)row * HDIM))[lane] = o;
}

// b_fused[o] = b_time[o] + sum_k cos(w2[k]*t2 + b2[k]) * W_time[o][128+k]
__global__ void k_timevec(const float* __restrict__ w2, const float* __restrict__ b2,
                          const float* __restrict__ W_time, const float* __restrict__ b_time,
                          float t2, float* __restrict__ b_fused)
{
  __shared__ float ht[HDIM];
  const int tid = threadIdx.x;          // blockDim.x == 128
  ht[tid] = cosf(w2[tid] * t2 + b2[tid]);
  __syncthreads();
  float s = b_time[tid];
  const float* wrow = W_time + (size_t)tid * 256 + 128;
  #pragma unroll 8
  for (int k = 0; k < HDIM; ++k) s += ht[k] * wrow[k];
  b_fused[tid] = s;
}

// per-relation sum of h[src] + edge counts (one wave per edge)
__global__ __launch_bounds__(256) void k_edge_rel(
    const int* __restrict__ src, const int* __restrict__ ety,
    const float* __restrict__ h, float* __restrict__ rel_sum,
    float* __restrict__ rel_cnt, int E)
{
  const int wave = threadIdx.x >> 5, lane = threadIdx.x & 31;
  const int e = blockIdx.x * 8 + wave;
  if (e >= E) return;
  const int s = src[e], r = ety[e];
  const float4 v = ((const float4*)(h + (size_t)s * HDIM))[lane];
  float* rp = rel_sum + (size_t)r * HDIM + lane * 4;
  atomicAdd(rp + 0, v.x); atomicAdd(rp + 1, v.y);
  atomicAdd(rp + 2, v.z); atomicAdd(rp + 3, v.w);
  if (lane == 0) atomicAdd(&rel_cnt[r], 1.0f);
}

__global__ void k_xinput(const float* __restrict__ rel_sum,
                         const float* __restrict__ rel_cnt,
                         float* __restrict__ x, int r2)
{
  int i = blockIdx.x * blockDim.x + threadIdx.x;
  if (i < r2 * HDIM) x[i] = rel_sum[i] / fmaxf(rel_cnt[i >> 7], 1.0f);
}

// agg[dst] += hW[src] + h0W[etype]; deg[dst] += 1   (one wave per edge)
__global__ __launch_bounds__(256) void k_edge_agg(
    const int* __restrict__ src, const int* __restrict__ dst,
    const int* __restrict__ ety,
    const float* __restrict__ hW, const float* __restrict__ h0W,
    float* __restrict__ agg, float* __restrict__ deg, int E)
{
  const int wave = threadIdx.x >> 5, lane = threadIdx.x & 31;
  const int e = blockIdx.x * 8 + wave;
  if (e >= E) return;
  const int s = src[e], d = dst[e], r = ety[e];
  const float4 a = ((const float4*)(hW  + (size_t)s * HDIM))[lane];
  const float4 b = ((const float4*)(h0W + (size_t)r * HDIM))[lane];
  float* ap = agg + (size_t)d * HDIM + lane * 4;
  atomicAdd(ap + 0, a.x + b.x); atomicAdd(ap + 1, a.y + b.y);
  atomicAdd(ap + 2, a.z + b.z); atomicAdd(ap + 3, a.w + b.w);
  if (lane == 0) atomicAdd(&deg[d], 1.0f);
}

// current_h = leaky(agg/max(deg,1) + hWloop)
__global__ void k_node(const float* __restrict__ agg, const float* __restrict__ deg,
                       const float* __restrict__ hWl, float* __restrict__ cur, int n)
{
  int i = blockIdx.x * blockDim.x + threadIdx.x;
  if (i < n * HDIM) {
    float v = agg[i] / fmaxf(deg[i >> 7], 1.0f) + hWl[i];
    cur[i] = v >= 0.f ? v : SLOPE_F * v;
  }
}

// h_out[row] = l2norm(GRU(gi[row], gh[row], h_prev[row])), gi/gh stride 384
__global__ __launch_bounds__(256) void k_gru(
    const float* __restrict__ gi, const float* __restrict__ gh,
    const float* __restrict__ hprev, float* __restrict__ out, int rows)
{
  const int wave = threadIdx.x >> 5, lane = threadIdx.x & 31;
  const int row = blockIdx.x * 8 + wave;
  if (row >= rows) return;
  const float4* gir = (const float4*)(gi + (size_t)row * 384);
  const float4* ghr = (const float4*)(gh + (size_t)row * 384);
  const float4 ir = gir[lane], iz = gir[32 + lane], innr = gir[64 + lane];
  const float4 hrr = ghr[lane], hz = ghr[32 + lane], hn = ghr[64 + lane];
  const float4 hp = ((const float4*)(hprev + (size_t)row * HDIM))[lane];

  float irv[4] = {ir.x, ir.y, ir.z, ir.w};
  float izv[4] = {iz.x, iz.y, iz.z, iz.w};
  float inv[4] = {innr.x, innr.y, innr.z, innr.w};
  float hrv[4] = {hrr.x, hrr.y, hrr.z, hrr.w};
  float hzv[4] = {hz.x, hz.y, hz.z, hz.w};
  float hnv[4] = {hn.x, hn.y, hn.z, hn.w};
  float hpv[4] = {hp.x, hp.y, hp.z, hp.w};

  float hv[4]; float ss = 0.f;
  #pragma unroll
  for (int c = 0; c < 4; ++c) {
    const float r = 1.f / (1.f + __expf(-(irv[c] + hrv[c])));
    const float z = 1.f / (1.f + __expf(-(izv[c] + hzv[c])));
    const float n = tanhf(inv[c] + r * hnv[c]);
    const float v = (1.f - z) * n + z * hpv[c];
    hv[c] = v; ss += v * v;
  }
  #pragma unroll
  for (int off = 16; off; off >>= 1) ss += __shfl_xor(ss, off, 32);
  const float s = 1.f / fmaxf(sqrtf(ss), 1e-12f);
  float4 o; o.x = hv[0] * s; o.y = hv[1] * s; o.z = hv[2] * s; o.w = hv[3] * s;
  ((float4*)(out + (size_t)row * HDIM))[lane] = o;
}

// ---------------------------------------------------------------------------
// Host orchestration
// ---------------------------------------------------------------------------
extern "C" void kernel_launch(void* const* d_in, const int* in_sizes, int n_in,
                              void* d_out, int out_size, void* d_ws, size_t ws_size,
                              hipStream_t stream) {
  const float* demb   = (const float*)d_in[0];
  const float* embrel = (const float*)d_in[1];
  const float* w_t2   = (const float*)d_in[2];
  const float* b_t2   = (const float*)d_in[3];
  const float* W_time = (const float*)d_in[4];
  const float* b_time = (const float*)d_in[5];
  const float* W_r    = (const float*)d_in[6];
  const float* b_r    = (const float*)d_in[7];
  const float* W_nei  = (const float*)d_in[8];
  const float* W_loop = (const float*)d_in[9];
  const float* Wih_e  = (const float*)d_in[10];
  const float* Whh_e  = (const float*)d_in[11];
  const float* bih_e  = (const float*)d_in[12];
  const float* bhh_e  = (const float*)d_in[13];
  const float* Wih_r  = (const float*)d_in[14];
  const float* Whh_r  = (const float*)d_in[15];
  const float* bih_r  = (const float*)d_in[16];
  const float* bhh_r  = (const float*)d_in[17];
  const int*   src    = (const int*)d_in[18];
  const int*   dstp   = (const int*)d_in[19];
  const int*   ety    = (const int*)d_in[20];

  const int H  = in_sizes[2];               // 128
  const int N  = in_sizes[0] / H;           // 50000
  const int R2 = in_sizes[1] / H;           // 400
  const int T  = 3;
  const int E  = in_sizes[18] / T;          // 500000

  // workspace carve-out
  float* w = (float*)d_ws;
  size_t off = 0;
  auto alloc = [&](size_t n) { float* p = w + off; off += n; return p; };
  float* hA      = alloc((size_t)N * H);
  float* hB      = alloc((size_t)N * H);
  float* h0A     = alloc((size_t)R2 * H);
  float* h0B     = alloc((size_t)R2 * H);
  float* b_fused = alloc(H);
  float* rel_sum = alloc((size_t)R2 * H);
  float* rel_cnt = alloc(R2);
  float* x_input = alloc((size_t)R2 * H);
  float* hWn     = alloc((size_t)N * H);
  float* hWl     = alloc((size_t)N * H);
  float* h0W     = alloc((size_t)R2 * H);
  float* agg     = alloc((size_t)N * H);
  float* deg     = alloc(N);
  float* cur     = alloc((size_t)N * H);
  float* gi      = alloc((size_t)N * 3 * H);
  float* gh      = alloc((size_t)N * 3 * H);
  float* gir     = alloc((size_t)R2 * 3 * H);
  float* ghr     = alloc((size_t)R2 * 3 * H);
  // packed bf16 weights (2 bf16 per float slot)
  auto balloc = [&](size_t n) { __bf16* p = (__bf16*)(w + off); off += (n + 1) / 2; return p; };
  __bf16* Wt_time = balloc((size_t)H * H);
  __bf16* Wt_r    = balloc((size_t)H * 2 * H);
  __bf16* Wt_nei  = balloc((size_t)H * H);
  __bf16* Wt_loop = balloc((size_t)H * H);
  __bf16* Wt_ihe  = balloc((size_t)3 * H * H);
  __bf16* Wt_hhe  = balloc((size_t)3 * H * H);
  __bf16* Wt_ihr  = balloc((size_t)3 * H * 2 * H);
  __bf16* Wt_hhr  = balloc((size_t)3 * H * H);
  (void)ws_size; (void)n_in; (void)out_size;

  auto prep = [&](const float* in, __bf16* out, int O, int K, int ld, int tr) {
    k_prep_w<<<dim3((O * K + 255) / 256), 256, 0, stream>>>(in, out, O, K, ld, tr);
  };
  // Pack all weights to bf16 [O][K] once (W_neigh / W_loop need transpose).
  prep(W_time, Wt_time, H, H, 2 * H, 0);        // first-half columns only
  prep(W_r,    Wt_r,    H, 2 * H, 2 * H, 0);
  prep(W_nei,  Wt_nei,  H, H, H, 1);
  prep(W_loop, Wt_loop, H, H, H, 1);
  prep(Wih_e,  Wt_ihe,  3 * H, H, H, 0);
  prep(Whh_e,  Wt_hhe,  3 * H, H, H, 0);
  prep(Wih_r,  Wt_ihr,  3 * H, 2 * H, 2 * H, 0);
  prep(Whh_r,  Wt_hhr,  3 * H, H, H, 0);

  auto gemm = [&](const float* A0, const float* A1, int K0, int lda0, int lda1,
                  const __bf16* Bt, const float* bias,
                  float* C, int ldc, int M, int O, int K) {
    dim3 grid((M / 16 + 3) / 4, O / 64);
    k_gemm_bf16<<<grid, 128, 0, stream>>>(A0, A1, K0, lda0, lda1,
                                          Bt, bias, C, ldc, M, O, K);
  };
  auto fill = [&](float* p, size_t n) {
    k_fill<<<dim3((unsigned)((n + 255) / 256)), 256, 0, stream>>>(p, (int)n, 0.0f);
  };

  // h = l2norm(dynamic_emb)
  k_l2norm<<<dim3((N + 7) / 8), 256, 0, stream>>>(demb, hA, N);
  float* h = hA;  float* hAlt = hB;
  float* h0 = h0A; float* h0Alt = h0B;

  for (int i = 0; i < T; ++i) {
    const float t2 = (float)(T - i + 1);

    // --- time-linear: h = cat(h, h_t) @ W_time.T + b_time
    //     broadcast-row half of the concat folded into b_fused
    k_timevec<<<1, 128, 0, stream>>>(w_t2, b_t2, W_time, b_time, t2, b_fused);
    gemm(h, nullptr, H, H, 0, Wt_time, b_fused, hAlt, H, N, H, H);
    { float* t = h; h = hAlt; hAlt = t; }

    const int* si = src + (size_t)i * E;
    const int* di = dstp + (size_t)i * E;
    const int* ei = ety + (size_t)i * E;

    // --- per-relation mean of h[src]  (only needed for h0 init / relation GRU)
    const bool needX = (i == 0) || (i < T - 1);
    if (needX) {
      fill(rel_sum, (size_t)R2 * H);
      fill(rel_cnt, R2);
      k_edge_rel<<<dim3((E + 7) / 8), 256, 0, stream>>>(si, ei, h, rel_sum, rel_cnt, E);
      k_xinput<<<dim3((R2 * H + 255) / 256), 256, 0, stream>>>(rel_sum, rel_cnt, x_input, R2);
    }
    if (i == 0) {
      // h0 = l2norm(cat(emb_rel, x_input) @ W_r.T + b_r)
      gemm(embrel, x_input, H, H, H, Wt_r, b_r, h0, H, R2, H, 2 * H);
      k_l2norm<<<dim3((R2 + 7) / 8), 256, 0, stream>>>(h0, h0, R2);
    }

    // --- factorized RGCN layer: (h[src]+h0[et]) @ W = (h@W)[src] + (h0@W)[et]
    gemm(h,  nullptr, H, H, 0, Wt_nei,  nullptr, hWn, H, N,  H, H);
    gemm(h0, nullptr, H, H, 0, Wt_nei,  nullptr, h0W, H, R2, H, H);
    gemm(h,  nullptr, H, H, 0, Wt_loop, nullptr, hWl, H, N,  H, H);
    fill(agg, (size_t)N * H);
    fill(deg, N);
    k_edge_agg<<<dim3((E + 7) / 8), 256, 0, stream>>>(si, di, ei, hWn, h0W, agg, deg, E);
    k_node<<<dim3((N * H + 255) / 256), 256, 0, stream>>>(agg, deg, hWl, cur, N);

    // --- entity GRU: gi = cur @ Wih_e.T + bih_e ; gh = h @ Whh_e.T + bhh_e
    gemm(cur, nullptr, H, H, 0, Wt_ihe, bih_e, gi, 3 * H, N, 3 * H, H);
    gemm(h,   nullptr, H, H, 0, Wt_hhe, bhh_e, gh, 3 * H, N, 3 * H, H);
    float* hout = (i == T - 1) ? (float*)d_out : hAlt;
    k_gru<<<dim3((N + 7) / 8), 256, 0, stream>>>(gi, gh, h, hout, N);
    if (i < T - 1) { float* t = h; h = hAlt; hAlt = t; }

    // --- relation GRU (result unused after the last step -> skip there)
    if (i < T - 1) {
      gemm(h0, x_input, H, H, H, Wt_ihr, bih_r, gir, 3 * H, R2, 3 * H, 2 * H);
      gemm(h0, nullptr, H, H, 0, Wt_hhr, bhh_r, ghr, 3 * H, R2, 3 * H, H);
      k_gru<<<dim3((R2 + 7) / 8), 256, 0, stream>>>(gir, ghr, h0, h0Alt, R2);
      { float* t = h0; h0 = h0Alt; h0Alt = t; }
    }
  }
}
